// BiLSTM_CRF_85100482003506
// MI455X (gfx1250) — compile-verified, matrix operand order checked by reading, not run on hardware
//
#include <hip/hip_runtime.h>

// ---------------------------------------------------------------------------
// BiLSTM-CRF for MI455X (gfx1250).  B=32 T=512 D_IN=1824 H=512 V=50000 K=30
// ---------------------------------------------------------------------------

typedef __attribute__((ext_vector_type(16))) __bf16 v16bf;
typedef __attribute__((ext_vector_type(8)))  float  v8f;

union Frag { uint4 u[2]; v16bf v; };

// ---- problem constants ----
#define NB    32
#define NT    512
#define DIN   1824
#define NH    512
#define G4H   2048
#define NK    30
#define KSTART 28
#define KSTOP  29
#define NEGV  (-10000.0f)
#define ROWS  (NT * NB)          // 16384 rows, row r = t*NB + b

// ---- workspace layout (bytes) ----
static constexpr size_t SZ_XB   = (size_t)ROWS * DIN * 2;      // bf16 gathered embeddings
static constexpr size_t SZ_WIHT = (size_t)G4H * DIN * 2;       // bf16 W_ih^T [2048,1824]
static constexpr size_t SZ_WHHT = (size_t)G4H * NH * 2;        // bf16 W_hh^T [2048,512]
static constexpr size_t SZ_XW   = (size_t)ROWS * G4H * 4;      // f32 input projections
static constexpr size_t SZ_H    = (size_t)ROWS * NH * 4;       // f32 hidden states

static constexpr size_t OFF_XB     = 0;
static constexpr size_t OFF_WIHT_F = OFF_XB + SZ_XB;
static constexpr size_t OFF_WIHT_B = OFF_WIHT_F + SZ_WIHT;
static constexpr size_t OFF_WHHT_F = OFF_WIHT_B + SZ_WIHT;
static constexpr size_t OFF_WHHT_B = OFF_WHHT_F + SZ_WHHT;
static constexpr size_t OFF_XW_F   = OFF_WHHT_B + SZ_WHHT;
static constexpr size_t OFF_XW_B   = OFF_XW_F + SZ_XW;
static constexpr size_t OFF_HF     = OFF_XW_B + SZ_XW;
static constexpr size_t OFF_HB     = OFF_HF + SZ_H;
static constexpr size_t OFF_FEATS  = OFF_HB + SZ_H;

// ---- helpers ----
__device__ __forceinline__ unsigned short f2bf(float f) {
  unsigned u = __float_as_uint(f);
  u = (u + 0x7FFFu + ((u >> 16) & 1u)) >> 16;       // round-to-nearest-even
  return (unsigned short)u;
}
__device__ __forceinline__ float sigm(float x) { return 1.f / (1.f + __expf(-x)); }
__device__ __forceinline__ float tanh_(float x) {
  x = fminf(fmaxf(x, -15.f), 15.f);
  float e = __expf(2.f * x);
  return (e - 1.f) / (e + 1.f);
}

// ---- Tensor Data Mover staging (CDNA5 TDM, 6-arg clang-23 form) ----
#if defined(__HIP_DEVICE_COMPILE__)
#if __has_builtin(__builtin_amdgcn_tensor_load_to_lds) && __has_builtin(__builtin_amdgcn_s_wait_tensorcnt)
#define HAVE_TDM 1
#endif
#endif

#ifdef HAVE_TDM
typedef __attribute__((ext_vector_type(4))) unsigned int v4u_;
typedef __attribute__((ext_vector_type(8))) int v8i_;
typedef __attribute__((ext_vector_type(4))) int v4i_;

// 2D tile load: tileD1 rows of tileD0 elements (2B each), row stride = strideElems.
__device__ __forceinline__ void tdm_load_2d(unsigned ldsOff, unsigned long long gaddr,
                                            unsigned tensorD0, unsigned tensorD1,
                                            unsigned tileD0, unsigned tileD1,
                                            unsigned long long strideElems) {
  v4u_ g0;
  g0.x = 1u;                                                   // count=1 valid D#
  g0.y = ldsOff;                                               // lds_addr
  g0.z = (unsigned)gaddr;                                      // global_addr[31:0]
  g0.w = ((unsigned)(gaddr >> 32) & 0x01FFFFFFu) | (2u << 30); // addr[56:32] | type=2
  v8i_ g1;
  g1[0] = (int)(1u << 16);                                     // data_size=1 (2 bytes)
  g1[1] = (int)((tensorD0 & 0xFFFFu) << 16);                   // tensor_dim0 lo
  g1[2] = (int)((tensorD0 >> 16) | ((tensorD1 & 0xFFFFu) << 16));
  g1[3] = (int)((tensorD1 >> 16) | (tileD0 << 16));
  g1[4] = (int)(tileD1 & 0xFFFFu);                             // tile_dim1, tile_dim2=0
  g1[5] = (int)(unsigned)(strideElems & 0xFFFFFFFFu);          // dim0 stride lo32
  g1[6] = (int)(unsigned)((strideElems >> 32) & 0xFFFFu);      // dim0 stride hi16
  g1[7] = 0;
  v4i_ z4 = {0, 0, 0, 0};
  v8i_ z8 = {0, 0, 0, 0, 0, 0, 0, 0};
  __builtin_amdgcn_tensor_load_to_lds(g0, g1, z4, z4, z8, 0);
}
#endif

// ===========================================================================
// K1: transpose f32 [Kd,N] -> bf16 [N,Kd]
// ===========================================================================
__global__ __launch_bounds__(256) void transpose_to_bf16(
    const float* __restrict__ in, unsigned short* __restrict__ out, int Kd, int N) {
  long idx = (long)blockIdx.x * 256 + threadIdx.x;
  long total = (long)Kd * N;
  if (idx < total) {
    int n = (int)(idx / Kd), k = (int)(idx % Kd);
    out[idx] = f2bf(in[(long)k * N + n]);
  }
}

// ===========================================================================
// K2: embedding gather -> bf16  xb[t*NB+b, d] = embed[ids[b,t], d]
// ===========================================================================
__global__ __launch_bounds__(256) void gather_embed(
    const int* __restrict__ ids, const float* __restrict__ embed,
    unsigned short* __restrict__ xb) {
  int r = blockIdx.x;                // 0..16383
  int t = r >> 5, b = r & 31;
  size_t src = (size_t)ids[b * NT + t] * DIN;
  size_t dst = (size_t)r * DIN;
  for (int d = threadIdx.x; d < DIN; d += 256)
    xb[dst + d] = f2bf(embed[src + d]);
}

// ===========================================================================
// K3: fused input-projection GEMM, bf16 WMMA, f32 accum, bias epilogue.
//     C[16384,2048] = A[16384,1824] x W^T[2048,1824]^T + bias
//     block tile 128x128, 8 waves (2M x 4N), wave tile 64x32, K-step 32.
// ===========================================================================
__global__ __launch_bounds__(256) void gemm_inproj(
    const unsigned short* __restrict__ A,
    const unsigned short* __restrict__ WTf, const unsigned short* __restrict__ WTb,
    const float* __restrict__ biasf, const float* __restrict__ biasb,
    float* __restrict__ Cf, float* __restrict__ Cb) {
  const unsigned short* W = blockIdx.z ? WTb : WTf;
  const float* bias = blockIdx.z ? biasb : biasf;
  float* C = blockIdx.z ? Cb : Cf;

  __shared__ unsigned short As[128 * 32];
  __shared__ unsigned short Bs[128 * 32];

  const int tid = threadIdx.x;
  const int lane = tid & 31, wave = tid >> 5;
  const int m16 = lane & 15, hi = lane >> 4;
  const int waveM = wave >> 2, waveN = wave & 3;
  const int m0 = blockIdx.y * 128;
  const int n0 = blockIdx.x * 128;

  const v8f vzero = {0, 0, 0, 0, 0, 0, 0, 0};
  v8f acc[4][2];
  for (int i = 0; i < 4; ++i)
    for (int j = 0; j < 2; ++j) acc[i][j] = vzero;

  const int srow = tid >> 1, shalf = tid & 1;   // manual-staging mapping

  for (int kt = 0; kt < DIN / 32; ++kt) {       // 57 K-steps
    const int k0 = kt * 32;
#ifdef HAVE_TDM
    if (wave == 0) {
      tdm_load_2d((unsigned)(size_t)(void*)As,
                  (unsigned long long)(size_t)(A + (size_t)m0 * DIN + k0),
                  DIN, ROWS, 32u, 128u, DIN);
      tdm_load_2d((unsigned)(size_t)(void*)Bs,
                  (unsigned long long)(size_t)(W + (size_t)n0 * DIN + k0),
                  DIN, G4H, 32u, 128u, DIN);
      __builtin_amdgcn_s_wait_tensorcnt(0);
    }
#else
    {
      const uint4* sa = (const uint4*)(A + (size_t)(m0 + srow) * DIN + k0) + shalf * 2;
      uint4* da = (uint4*)(&As[srow * 32]) + shalf * 2;
      da[0] = sa[0]; da[1] = sa[1];
      const uint4* sb = (const uint4*)(W + (size_t)(n0 + srow) * DIN + k0) + shalf * 2;
      uint4* db = (uint4*)(&Bs[srow * 32]) + shalf * 2;
      db[0] = sb[0]; db[1] = sb[1];
      if (kt + 1 < DIN / 32) {
        __builtin_prefetch(A + (size_t)(m0 + srow) * DIN + k0 + 32);
        __builtin_prefetch(W + (size_t)(n0 + srow) * DIN + k0 + 32);
      }
    }
#endif
    __syncthreads();

    Frag bfrag[2];
    for (int j = 0; j < 2; ++j) {
      const unsigned short* p = &Bs[(waveN * 32 + j * 16 + m16) * 32 + hi * 8];
      bfrag[j].u[0] = *(const uint4*)p;
      bfrag[j].u[1] = *(const uint4*)(p + 16);
    }
    for (int i = 0; i < 4; ++i) {
      Frag a;
      const unsigned short* p = &As[(waveM * 64 + i * 16 + m16) * 32 + hi * 8];
      a.u[0] = *(const uint4*)p;
      a.u[1] = *(const uint4*)(p + 16);
      for (int j = 0; j < 2; ++j)
        acc[i][j] = __builtin_amdgcn_wmma_f32_16x16x32_bf16(
            false, a.v, false, bfrag[j].v, (short)0, acc[i][j], false, false);
    }
    __syncthreads();
  }

  for (int i = 0; i < 4; ++i)
    for (int j = 0; j < 2; ++j) {
      int col = n0 + waveN * 32 + j * 16 + m16;
      float bj = bias[col];
      for (int v = 0; v < 8; ++v) {
        int row = m0 + waveM * 64 + i * 16 + v + hi * 8;
        C[(size_t)row * G4H + col] = acc[i][j][v] + bj;
      }
    }
}

// ===========================================================================
// K4: LSTM scan.  grid = (2 batch-halves, 2 directions), 256 threads.
//     Per step: gates[16,2048] = h_bf16[16,512] x Whh^T + xw  (WMMA),
//     then elementwise cell update.  h (bf16, double-buffered) and c (f32)
//     live in LDS (64 KB).  Each wave owns 64 columns x all 4 gates.
// ===========================================================================
__global__ __launch_bounds__(256) void lstm_scan(
    const float* __restrict__ xw_f, const float* __restrict__ xw_b,
    const unsigned short* __restrict__ WTf, const unsigned short* __restrict__ WTb,
    float* __restrict__ hf, float* __restrict__ hb) {
  const int dir = blockIdx.y;
  const float* xw = dir ? xw_b : xw_f;
  const unsigned short* W = dir ? WTb : WTf;
  float* hout = dir ? hb : hf;
  const int b0 = blockIdx.x * 16;

  __shared__ unsigned short hbf[2][16 * NH];  // 2 x 16 KB
  __shared__ float cst[16 * NH];              // 32 KB

  const int tid = threadIdx.x;
  for (int i = tid; i < 16 * NH; i += 256) { hbf[0][i] = 0; cst[i] = 0.f; }
  __syncthreads();

  const int lane = tid & 31, wave = tid >> 5;
  const int m16 = lane & 15, hi = lane >> 4;
  const v8f vzero = {0, 0, 0, 0, 0, 0, 0, 0};

  for (int s = 0; s < NT; ++s) {
    const int t = dir ? (NT - 1 - s) : s;
    const int cur = s & 1;

    for (int ct = 0; ct < 4; ++ct) {
      const int jbase = wave * 64 + ct * 16;       // column tile base (0..511)
      v8f aI = vzero, aF = vzero, aG = vzero, aO = vzero;

      for (int ks = 0; ks < NH / 32; ++ks) {       // 16 K-steps
        Frag a;
        const unsigned short* ap = &hbf[cur][m16 * NH + ks * 32 + hi * 8];
        a.u[0] = *(const uint4*)ap;
        a.u[1] = *(const uint4*)(ap + 16);

        const size_t kcol = (size_t)(ks * 32 + hi * 8);
        Frag bI, bF, bG, bO;
        const unsigned short* w0 = W + (size_t)(0 * NH + jbase + m16) * NH + kcol;
        const unsigned short* w1 = W + (size_t)(1 * NH + jbase + m16) * NH + kcol;
        const unsigned short* w2 = W + (size_t)(2 * NH + jbase + m16) * NH + kcol;
        const unsigned short* w3 = W + (size_t)(3 * NH + jbase + m16) * NH + kcol;
        bI.u[0] = *(const uint4*)w0; bI.u[1] = *(const uint4*)(w0 + 16);
        bF.u[0] = *(const uint4*)w1; bF.u[1] = *(const uint4*)(w1 + 16);
        bG.u[0] = *(const uint4*)w2; bG.u[1] = *(const uint4*)(w2 + 16);
        bO.u[0] = *(const uint4*)w3; bO.u[1] = *(const uint4*)(w3 + 16);

        aI = __builtin_amdgcn_wmma_f32_16x16x32_bf16(false, a.v, false, bI.v, (short)0, aI, false, false);
        aF = __builtin_amdgcn_wmma_f32_16x16x32_bf16(false, a.v, false, bF.v, (short)0, aF, false, false);
        aG = __builtin_amdgcn_wmma_f32_16x16x32_bf16(false, a.v, false, bG.v, (short)0, aG, false, false);
        aO = __builtin_amdgcn_wmma_f32_16x16x32_bf16(false, a.v, false, bO.v, (short)0, aO, false, false);
      }

      const int jc = jbase + m16;
      for (int v = 0; v < 8; ++v) {
        const int m = v + hi * 8;
        const float* xr = xw + ((size_t)t * NB + (b0 + m)) * G4H;
        float gi = aI[v] + xr[jc];
        float gf = aF[v] + xr[NH + jc];
        float gg = aG[v] + xr[2 * NH + jc];
        float go = aO[v] + xr[3 * NH + jc];
        float c = sigm(gf) * cst[m * NH + jc] + sigm(gi) * tanh_(gg);
        float h = sigm(go) * tanh_(c);
        cst[m * NH + jc] = c;
        hbf[cur ^ 1][m * NH + jc] = f2bf(h);
        hout[((size_t)t * NB + (b0 + m)) * NH + jc] = h;
      }
    }
    __syncthreads();
  }
}

// ===========================================================================
// K5: tag projection  feats[b,t,k] = [hf|hb] . W_tag[:,k] + b_tag[k]
// ===========================================================================
__global__ __launch_bounds__(32) void tag_proj(
    const float* __restrict__ hf, const float* __restrict__ hb,
    const float* __restrict__ Wt, const float* __restrict__ bt,
    float* __restrict__ feats) {
  __shared__ float hrow[2 * NH];
  int r = blockIdx.x;                       // t*NB + b
  const float* hfr = hf + (size_t)r * NH;
  const float* hbr = hb + (size_t)r * NH;
  for (int d = threadIdx.x; d < NH; d += 32) {
    hrow[d] = hfr[d];
    hrow[NH + d] = hbr[d];
  }
  __syncthreads();
  int k = threadIdx.x;
  if (k < NK) {
    float acc = bt[k];
    for (int d = 0; d < 2 * NH; ++d) acc += hrow[d] * Wt[d * NK + k];
    int t = r >> 5, b = r & 31;
    feats[((size_t)b * NT + t) * NK + k] = acc;
  }
}

// ===========================================================================
// K6: CRF negative log-likelihood, one block (one wave) per batch element.
// ===========================================================================
__global__ __launch_bounds__(32) void crf_nll(
    const float* __restrict__ feats, const int* __restrict__ tags,
    const float* __restrict__ transitions, float* __restrict__ out) {
  __shared__ float tr[NK * NK];
  __shared__ float alpha[2][32];
  __shared__ float red[32];

  const int b = blockIdx.x, lane = threadIdx.x;
  for (int i = lane; i < NK * NK; i += 32) tr[i] = transitions[i];
  alpha[0][lane] = (lane == KSTART) ? 0.f : NEGV;
  __syncthreads();

  const float* fb = feats + (size_t)b * NT * NK;
  int cur = 0;
  for (int t = 0; t < NT; ++t) {
    float na = NEGV;
    if (lane < NK) {
      float m = -3.0e38f;
      for (int p = 0; p < NK; ++p) m = fmaxf(m, alpha[cur][p] + tr[lane * NK + p]);
      float ssum = 0.f;
      for (int p = 0; p < NK; ++p) ssum += __expf(alpha[cur][p] + tr[lane * NK + p] - m);
      na = m + __logf(ssum) + fb[t * NK + lane];
    }
    alpha[cur ^ 1][lane] = na;
    __syncthreads();
    cur ^= 1;
  }

  red[lane] = (lane < NK) ? alpha[cur][lane] + tr[KSTOP * NK + lane] : -3.0e38f;
  __syncthreads();
  if (lane == 0) {
    float m = -3.0e38f;
    for (int p = 0; p < NK; ++p) m = fmaxf(m, red[p]);
    float ssum = 0.f;
    for (int p = 0; p < NK; ++p) ssum += __expf(red[p] - m);
    float logZ = m + __logf(ssum);
    float gold = 0.f;
    int prev = KSTART;
    for (int t = 0; t < NT; ++t) {
      int nt = tags[b * NT + t];
      gold += tr[nt * NK + prev] + fb[t * NK + nt];
      prev = nt;
    }
    gold += tr[KSTOP * NK + prev];
    out[b] = logZ - gold;
  }
}

// ===========================================================================
extern "C" void kernel_launch(void* const* d_in, const int* in_sizes, int n_in,
                              void* d_out, int out_size, void* d_ws, size_t ws_size,
                              hipStream_t stream) {
  (void)in_sizes; (void)n_in; (void)out_size; (void)ws_size;
  const int*   ids   = (const int*)d_in[0];
  const int*   tags  = (const int*)d_in[1];
  const float* embed = (const float*)d_in[2];
  const float* Wihf  = (const float*)d_in[3];
  const float* Whhf  = (const float*)d_in[4];
  const float* bfv   = (const float*)d_in[5];
  const float* Wihb  = (const float*)d_in[6];
  const float* Whhb  = (const float*)d_in[7];
  const float* bbv   = (const float*)d_in[8];
  const float* Wtag  = (const float*)d_in[9];
  const float* btag  = (const float*)d_in[10];
  const float* trans = (const float*)d_in[11];
  float* out = (float*)d_out;
  char*  ws  = (char*)d_ws;

  unsigned short* xb    = (unsigned short*)(ws + OFF_XB);
  unsigned short* wihtf = (unsigned short*)(ws + OFF_WIHT_F);
  unsigned short* wihtb = (unsigned short*)(ws + OFF_WIHT_B);
  unsigned short* whhtf = (unsigned short*)(ws + OFF_WHHT_F);
  unsigned short* whhtb = (unsigned short*)(ws + OFF_WHHT_B);
  float* xwf   = (float*)(ws + OFF_XW_F);
  float* xwb   = (float*)(ws + OFF_XW_B);
  float* hfp   = (float*)(ws + OFF_HF);
  float* hbp   = (float*)(ws + OFF_HB);
  float* featp = (float*)(ws + OFF_FEATS);

  // 1) weight transpose + bf16 convert
  transpose_to_bf16<<<(DIN * G4H + 255) / 256, 256, 0, stream>>>(Wihf, wihtf, DIN, G4H);
  transpose_to_bf16<<<(DIN * G4H + 255) / 256, 256, 0, stream>>>(Wihb, wihtb, DIN, G4H);
  transpose_to_bf16<<<(NH * G4H + 255) / 256, 256, 0, stream>>>(Whhf, whhtf, NH, G4H);
  transpose_to_bf16<<<(NH * G4H + 255) / 256, 256, 0, stream>>>(Whhb, whhtb, NH, G4H);

  // 2) embedding gather -> bf16
  gather_embed<<<ROWS, 256, 0, stream>>>(ids, embed, xb);

  // 3) big WMMA GEMM: both directions (grid.z)
  gemm_inproj<<<dim3(G4H / 128, ROWS / 128, 2), 256, 0, stream>>>(
      xb, wihtf, wihtb, bfv, bbv, xwf, xwb);

  // 4) LSTM scans (2 batch halves x 2 directions)
  lstm_scan<<<dim3(2, 2), 256, 0, stream>>>(xwf, xwb, whhtf, whhtb, hfp, hbp);

  // 5) tag projection
  tag_proj<<<ROWS, 32, 0, stream>>>(hfp, hbp, Wtag, btag, featp);

  // 6) CRF NLL per batch element
  crf_nll<<<32, 32, 0, stream>>>(featp, tags, trans, out);
}
